// Masker_28166395527230
// MI455X (gfx1250) — compile-verified, matrix operand order checked by reading, not run on hardware
//
#include <hip/hip_runtime.h>
#include <hip/hip_bf16.h>

#define BB 16
#define NN 16
#define HH 640
#define WW 640
#define PHH 300
#define PWW 300
#define MIN_AREA_F 60.0f
#define MAX_ANG_F 0.34906584f   // 20 deg in radians

typedef __attribute__((ext_vector_type(16))) _Float16 v16h;
typedef __attribute__((ext_vector_type(8)))  float    v8f;
typedef __attribute__((ext_vector_type(4)))  unsigned u32x4;
typedef __attribute__((ext_vector_type(8)))  int      i32x8;
typedef __attribute__((ext_vector_type(4)))  int      i32x4;

// ---------------- threefry2x32 (20 rounds, JAX rotation schedule) ----------------
struct U2 { unsigned a, b; };

__device__ __forceinline__ unsigned rotl32(unsigned x, int r) {
    return (x << r) | (x >> (32 - r));
}

__device__ __forceinline__ U2 tf2x32(U2 k, U2 c) {
    const unsigned ks0 = k.a, ks1 = k.b, ks2 = 0x1BD11BDAu ^ k.a ^ k.b;
    unsigned x0 = c.a + ks0, x1 = c.b + ks1;
    const int RA[4] = {13, 15, 26, 6};
    const int RB[4] = {17, 29, 16, 24};
#pragma unroll
    for (int g = 0; g < 5; ++g) {
        const int* R = (g & 1) ? RB : RA;
#pragma unroll
        for (int j = 0; j < 4; ++j) { x0 += x1; x1 = rotl32(x1, R[j]); x1 ^= x0; }
        switch (g) {
            case 0: x0 += ks1; x1 += ks2 + 1u; break;
            case 1: x0 += ks2; x1 += ks0 + 2u; break;
            case 2: x0 += ks0; x1 += ks1 + 3u; break;
            case 3: x0 += ks1; x1 += ks2 + 4u; break;
            case 4: x0 += ks2; x1 += ks0 + 5u; break;
        }
    }
    U2 r; r.a = x0; r.b = x1; return r;
}

__device__ __forceinline__ U2 ksplit(U2 k, unsigned i) { U2 c; c.a = 0u; c.b = i; return tf2x32(k, c); }

__device__ __forceinline__ float u01(unsigned bits) {
    return __uint_as_float((bits >> 9) | 0x3F800000u) - 1.0f;   // [0,1)
}

__device__ __forceinline__ float uni1(U2 k, float lo, float hi) {
    U2 c; c.a = 0u; c.b = 0u;
    U2 r = tf2x32(k, c);
    return lo + (hi - lo) * u01(r.a);
}

__device__ __forceinline__ void normal2(U2 k, unsigned ctr, float* z0, float* z1) {
    U2 c; c.a = 0u; c.b = ctr;
    U2 r = tf2x32(k, c);
    float u1 = 1.0f - u01(r.a);                 // (0,1]
    float u2 = u01(r.b);
    float rad = sqrtf(-2.0f * logf(u1));
    float s, co; sincosf(6.28318530718f * u2, &s, &co);
    *z0 = rad * co; *z1 = rad * s;
}

__device__ __forceinline__ float clamp11(float v) { return fminf(fmaxf(v, -1.0f), 1.0f); }

// jittered + brightness-shifted patch value (folded, never materialized)
__device__ __forceinline__ float ptx(float v, float wc, float bc, float d) {
    return clamp11(clamp11(wc * v + bc) + d);
}

// ---------------- kernel 0: zero workspace accumulators ----------------
__global__ void zero_ws_kernel(float* __restrict__ ws) {
    int i = threadIdx.x;
    if (i < BB * 4) ws[i] = 0.0f;
}

// ---------------- kernel 1: per-image means via WMMA reduction ----------------
// ws[b*4+0] = sum(images[b]) ; ws[b*4+1] = sum(clip(w*patch+b)) with per-image w,b
__global__ __launch_bounds__(256) void means_kernel(const float* __restrict__ images,
                                                    const float* __restrict__ patch,
                                                    float* __restrict__ ws) {
    const int b    = blockIdx.x;
    const int tid  = threadIdx.x;
    const int wave = tid >> 5;
    const int lane = tid & 31;
    const int gw   = blockIdx.y * 8 + wave;     // 0..63 waves per image

    // per-image patch jitter params (every thread recomputes; cheap, scalarizes)
    U2 base; base.a = 0u; base.b = 42u;
    U2 kimg = ksplit(base, (unsigned)b);
    U2 kw = ksplit(kimg, 0u);
    U2 kb = ksplit(kimg, 1u);
    float z0, z1, z2, zz;
    normal2(kw, 0u, &z0, &z1); normal2(kw, 1u, &z2, &zz);
    const float wv0 = z0 * 0.1f + 0.5f, wv1 = z1 * 0.1f + 0.5f, wv2 = z2 * 0.1f + 0.5f;
    normal2(kb, 0u, &z0, &z1); normal2(kb, 1u, &z2, &zz);
    const float bv0 = z0 * 0.01f, bv1 = z1 * 0.01f, bv2 = z2 * 0.01f;
    const float WV[3] = {wv0, wv1, wv2};
    const float BV[3] = {bv0, bv1, bv2};

    v16h ones;
#pragma unroll
    for (int j = 0; j < 16; ++j) ones[j] = (_Float16)1.0f;

    // ---- image sum: 512 contiguous floats per wave per WMMA ----
    const float* img = images + (size_t)b * HH * WW * 3;
    const int NT = HH * WW * 3;                 // 1,228,800 : divisible by 512
    v8f acc = {0.f, 0.f, 0.f, 0.f, 0.f, 0.f, 0.f, 0.f};
    for (int bas = gw * 512; bas < NT; bas += 64 * 512) {
        const int off = bas + lane * 16;
        const float4* q = (const float4*)(img + off);
        float4 q0 = q[0], q1 = q[1], q2 = q[2], q3 = q[3];
        v16h a;
        a[0]  = (_Float16)q0.x; a[1]  = (_Float16)q0.y; a[2]  = (_Float16)q0.z; a[3]  = (_Float16)q0.w;
        a[4]  = (_Float16)q1.x; a[5]  = (_Float16)q1.y; a[6]  = (_Float16)q1.z; a[7]  = (_Float16)q1.w;
        a[8]  = (_Float16)q2.x; a[9]  = (_Float16)q2.y; a[10] = (_Float16)q2.z; a[11] = (_Float16)q2.w;
        a[12] = (_Float16)q3.x; a[13] = (_Float16)q3.y; a[14] = (_Float16)q3.z; a[15] = (_Float16)q3.w;
        acc = __builtin_amdgcn_wmma_f32_16x16x32_f16(false, a, false, ones, (short)0, acc,
                                                     false, false);
    }
    float s = acc[0] + acc[1] + acc[2] + acc[3] + acc[4] + acc[5] + acc[6] + acc[7];
    s += __shfl_down(s, 16, 32);                // rows 0..7 (lane0) + rows 8..15 (lane16)
    if (lane == 0) atomicAdd(&ws[b * 4 + 0], s);

    // ---- jittered-patch sum ----
    const int NP = PHH * PWW * 3;               // 270,000 : has a 512-tail
    v8f acc2 = {0.f, 0.f, 0.f, 0.f, 0.f, 0.f, 0.f, 0.f};
    for (int bas = gw * 512; bas < NP; bas += 64 * 512) {
        const int off = bas + lane * 16;
        v16h a;
#pragma unroll
        for (int j = 0; j < 16; ++j) {
            const int e  = off + j;
            const int ec = e < NP ? e : NP - 1;
            const int ch = ec % 3;
            const float t = clamp11(WV[ch] * patch[ec] + BV[ch]);
            a[j] = (_Float16)((e < NP) ? t : 0.0f);
        }
        acc2 = __builtin_amdgcn_wmma_f32_16x16x32_f16(false, a, false, ones, (short)0, acc2,
                                                      false, false);
    }
    float s2 = acc2[0] + acc2[1] + acc2[2] + acc2[3] + acc2[4] + acc2[5] + acc2[6] + acc2[7];
    s2 += __shfl_down(s2, 16, 32);
    if (lane == 0) atomicAdd(&ws[b * 4 + 1], s2);
}

// ---------------- kernel 2: fused per-pixel sequential paste ----------------
struct BoxP {
    float y0i, x0i, diagi, phi, cc, ca, sa, top, bright, valid;
    unsigned k2a, k2b;
};

__global__ __launch_bounds__(256) void apply_kernel(const float* __restrict__ boxes,
                                                    const float* __restrict__ images,
                                                    const float* __restrict__ patch,
                                                    const float* __restrict__ scalep,
                                                    const float* __restrict__ ws,
                                                    float* __restrict__ out) {
    __shared__ float sbox[NN * 4];              // staged via Tensor Data Mover
    __shared__ BoxP sp[NN];
    __shared__ float swv[3], sbv[3], sdelta;

    const int b = blockIdx.z;
    const int x = blockIdx.x * 16 + threadIdx.x;
    const int y = blockIdx.y * 16 + threadIdx.y;
    const int tid = threadIdx.y * 16 + threadIdx.x;

    // ---- stage this image's box table (256 B) into LDS ----
#if __has_builtin(__builtin_amdgcn_tensor_load_to_lds)
    if (tid < 32) {                              // wave 0 only (wave-uniform branch)
        const unsigned lds_off = (unsigned)(size_t)&sbox[0];        // addr[31:0] = LDS offset
        const unsigned long long ga =
            (unsigned long long)(size_t)(boxes + (size_t)b * NN * 4);
        u32x4 g0;
        g0[0] = 1u;                              // count=1, user descriptor
        g0[1] = lds_off;                         // lds_addr
        g0[2] = (unsigned)(ga & 0xFFFFFFFFull);  // global_addr[31:0]
        g0[3] = (unsigned)((ga >> 32) & 0x1FFFFFFull) | (2u << 30); // addr[56:32] | type=2
        i32x8 g1;
        g1[0] = 0x20000;                         // data_size=2 (4 B), no multicast/pad/iterate
        g1[1] = (int)(64u << 16);                // tensor_dim0 = 64 (low 16 into [31:16])
        g1[2] = (int)(1u << 16);                 // tensor_dim0 hi=0 ; tensor_dim1 = 1
        g1[3] = (int)(64u << 16);                // tensor_dim1 hi=0 ; tile_dim0 = 64
        g1[4] = 1;                               // tile_dim1 = 1 ; tile_dim2 = 0
        g1[5] = 64;                              // tensor_dim0_stride = 64
        g1[6] = 0;
        g1[7] = 0;
        i32x4 zz4 = {0, 0, 0, 0};
#if __clang_major__ >= 23
        i32x8 zz8 = {0, 0, 0, 0, 0, 0, 0, 0};
        __builtin_amdgcn_tensor_load_to_lds(g0, g1, zz4, zz4, zz8, 0);
#else
        __builtin_amdgcn_tensor_load_to_lds(g0, g1, zz4, zz4, 0);
#endif
        __builtin_amdgcn_s_wait_tensorcnt(0);    // TENSORcnt==0 : tile landed in LDS
    }
#else
    if (tid < NN * 4) sbox[tid] = boxes[(size_t)b * NN * 4 + tid];
#endif
    __syncthreads();

    if (tid < NN) {
        const int n = tid;
        U2 base; base.a = 0u; base.b = 42u;
        U2 kimg  = ksplit(base, (unsigned)b);
        U2 kloop = ksplit(kimg, 2u);
        U2 kk = ksplit(kloop, (unsigned)n);
        U2 k1 = ksplit(kk, 0u);
        U2 k2 = ksplit(kk, 1u);
        U2 k3 = ksplit(kk, 2u);
        const float scale = scalep[0];
        const float ymin = sbox[n * 4 + 0], xmin = sbox[n * 4 + 1];
        const float ymax = sbox[n * 4 + 2], xmax = sbox[n * 4 + 3];
        const float h = ymax - ymin, wd = xmax - xmin;
        const float longer = fmaxf(h, wd);
        const float ps = floorf(longer * scale);
        const float diag = fminf(1.41421356237f * ps, (float)WW);
        const float oy = ymin + h * 0.5f, ox = xmin + wd * 0.5f;
        float y0 = fmaxf(oy - diag * 0.5f, 0.0f);
        float x0 = fmaxf(ox - diag * 0.5f, 0.0f);
        y0 = (y0 + diag > (float)HH) ? (float)HH - diag : y0;
        x0 = (x0 + diag > (float)WW) ? (float)WW - diag : x0;
        BoxP p;
        p.y0i = floorf(y0); p.x0i = floorf(x0);
        p.phi = fmaxf(floorf(ps), 1.0f);
        p.diagi = floorf(diag);
        p.valid = (p.phi * p.phi > MIN_AREA_F) ? 1.0f : 0.0f;
        const float ang = uni1(k1, -MAX_ANG_F, MAX_ANG_F);
        float sna, csa; sincosf(ang, &sna, &csa);
        p.ca = csa; p.sa = sna;
        p.top = floorf((p.diagi - p.phi) * 0.5f);
        p.cc = (p.diagi - 1.0f) * 0.5f;
        p.bright = uni1(k3, -0.3f, 0.3f);
        p.k2a = k2.a; p.k2b = k2.b;
        sp[n] = p;
        // warm L2/WGP$ for the patch region this block may sample (global_prefetch_b8)
        __builtin_prefetch(patch + (size_t)n * (PHH / NN) * PWW * 3, 0, 3);
    } else if (tid == NN) {
        U2 base; base.a = 0u; base.b = 42u;
        U2 kimg = ksplit(base, (unsigned)b);
        U2 kw = ksplit(kimg, 0u);
        U2 kb = ksplit(kimg, 1u);
        float z0, z1, z2, zz;
        normal2(kw, 0u, &z0, &z1); normal2(kw, 1u, &z2, &zz);
        swv[0] = z0 * 0.1f + 0.5f; swv[1] = z1 * 0.1f + 0.5f; swv[2] = z2 * 0.1f + 0.5f;
        normal2(kb, 0u, &z0, &z1); normal2(kb, 1u, &z2, &zz);
        sbv[0] = z0 * 0.01f; sbv[1] = z1 * 0.01f; sbv[2] = z2 * 0.01f;
        sdelta = ws[b * 4 + 0] * (1.0f / (float)(HH * WW * 3)) -
                 ws[b * 4 + 1] * (1.0f / (float)(PHH * PWW * 3));
    }
    __syncthreads();

    const size_t pix = ((size_t)b * HH + y) * WW + x;
    const float* ip = images + pix * 3;
    const float o0 = ip[0], o1 = ip[1], o2 = ip[2];
    float c0 = o0, c1 = o1, c2 = o2;           // current background (evolves over boxes)
    float m0 = 0.f, m1 = 0.f, m2 = 0.f;        // mask
    const float fy = (float)y, fx = (float)x;
    const float dv = sdelta;
    const float w0 = swv[0], w1 = swv[1], w2 = swv[2];
    const float bb0 = sbv[0], bb1 = sbv[1], bb2 = sbv[2];
    const unsigned pixloc = (unsigned)(y * WW + x);

    for (int n = 0; n < NN; ++n) {
        const BoxP p = sp[n];
        if (p.valid == 0.0f) continue;
        const float u = fy - p.y0i, v = fx - p.x0i;
        if (!(u >= 0.f && v >= 0.f && u < p.diagi && v < p.diagi)) continue;   // in_sq
        const float su = p.cc + p.ca * (u - p.cc) - p.sa * (v - p.cc);
        const float sv = p.cc + p.sa * (u - p.cc) + p.ca * (v - p.cc);
        const float py = su - p.top, px = sv - p.top;
        const bool vs = (py >= 0.f) && (py <= p.phi - 1.f) &&
                        (px >= 0.f) && (px <= p.phi - 1.f);
        if (vs) {
            const float sy = (py + 0.5f) * ((float)PHH / p.phi) - 0.5f;
            const float sx = (px + 0.5f) * ((float)PWW / p.phi) - 0.5f;
            const float fy0 = floorf(sy), fx0 = floorf(sx);
            const float wy = sy - fy0, wx = sx - fx0;
            int yi0 = (int)fy0; yi0 = yi0 < 0 ? 0 : (yi0 > PHH - 1 ? PHH - 1 : yi0);
            int xi0 = (int)fx0; xi0 = xi0 < 0 ? 0 : (xi0 > PWW - 1 ? PWW - 1 : xi0);
            const int yi1 = yi0 + 1 > PHH - 1 ? PHH - 1 : yi0 + 1;
            const int xi1 = xi0 + 1 > PWW - 1 ? PWW - 1 : xi0 + 1;
            const float* t00 = patch + ((size_t)yi0 * PWW + xi0) * 3;
            const float* t01 = patch + ((size_t)yi0 * PWW + xi1) * 3;
            const float* t10 = patch + ((size_t)yi1 * PWW + xi0) * 3;
            const float* t11 = patch + ((size_t)yi1 * PWW + xi1) * 3;
            const float g00 = (1.f - wy) * (1.f - wx), g01 = (1.f - wy) * wx;
            const float g10 = wy * (1.f - wx),         g11 = wy * wx;
            const float s0 = ptx(t00[0], w0, bb0, dv) * g00 + ptx(t01[0], w0, bb0, dv) * g01 +
                             ptx(t10[0], w0, bb0, dv) * g10 + ptx(t11[0], w0, bb0, dv) * g11;
            const float s1 = ptx(t00[1], w1, bb1, dv) * g00 + ptx(t01[1], w1, bb1, dv) * g01 +
                             ptx(t10[1], w1, bb1, dv) * g10 + ptx(t11[1], w1, bb1, dv) * g11;
            const float s2 = ptx(t00[2], w2, bb2, dv) * g00 + ptx(t01[2], w2, bb2, dv) * g01 +
                             ptx(t10[2], w2, bb2, dv) * g10 + ptx(t11[2], w2, bb2, dv) * g11;
            U2 k2; k2.a = p.k2a; k2.b = p.k2b;
            U2 cc0; cc0.a = pixloc; cc0.b = 0u;
            U2 cc1; cc1.a = pixloc; cc1.b = 1u;
            const U2 r0 = tf2x32(k2, cc0);
            const U2 r1 = tf2x32(k2, cc1);
            const float n0 = u01(r0.a) * 0.2f - 0.1f;
            const float n1 = u01(r0.b) * 0.2f - 0.1f;
            const float n2 = u01(r1.a) * 0.2f - 0.1f;
            const float i0 = clamp11(s0 + n0 + p.bright);
            const float i1 = clamp11(s1 + n1 + p.bright);
            const float i2 = clamp11(s2 + n2 + p.bright);
            c0 = i0; c1 = i1; c2 = i2;
            m0 = o0 - i0; m1 = o1 - i1; m2 = o2 - i2;
        } else {
            // inside square but patch-invalid: image unchanged, mask = orig - current bg
            m0 = o0 - c0; m1 = o1 - c1; m2 = o2 - c2;
        }
    }

    float* oi = out + pix * 3;
    float* om = out + (size_t)BB * HH * WW * 3 + pix * 3;
    oi[0] = c0; oi[1] = c1; oi[2] = c2;
    om[0] = m0; om[1] = m1; om[2] = m2;
}

extern "C" void kernel_launch(void* const* d_in, const int* in_sizes, int n_in,
                              void* d_out, int out_size, void* d_ws, size_t ws_size,
                              hipStream_t stream) {
    const float* boxes  = (const float*)d_in[0];   // [B,N,4]
    const float* images = (const float*)d_in[1];   // [B,H,W,3]
    const float* patch  = (const float*)d_in[2];   // [PH,PW,3]
    const float* scale  = (const float*)d_in[3];   // scalar
    float* out = (float*)d_out;                    // img_out ++ mask_out
    float* ws  = (float*)d_ws;                     // B*4 accumulators

    zero_ws_kernel<<<1, 64, 0, stream>>>(ws);
    means_kernel<<<dim3(BB, 8), 256, 0, stream>>>(images, patch, ws);
    apply_kernel<<<dim3(WW / 16, HH / 16, BB), dim3(16, 16), 0, stream>>>(
        boxes, images, patch, scale, ws, out);
}
